// LinearAttention_8839042695251
// MI455X (gfx1250) — matrix operand order repaired and checked by reading, not compiled, not run on hardware
//
#include <hip/hip_runtime.h>
#include <hip/hip_bf16.h>

typedef __attribute__((ext_vector_type(16))) __bf16 v16bf;
typedef __attribute__((ext_vector_type(8)))  float  v8f;

union Frag16 { v16bf v; unsigned u[8]; };

__device__ __forceinline__ unsigned short f2bf(float f) {
  unsigned u = __builtin_bit_cast(unsigned, f);
  u += 0x7FFFu + ((u >> 16) & 1u);           // round-to-nearest-even
  return (unsigned short)(u >> 16);
}
// store 8 u16 (one uint4) to LDS as 4 dword stores (rows only 4B-aligned)
__device__ __forceinline__ void st8(unsigned short* dst, uint4 x) {
  unsigned* d = (unsigned*)dst;
  d[0] = x.x; d[1] = x.y; d[2] = x.z; d[3] = x.w;
}
// 16-bit A-matrix 16x32 fragment K-mapping (contiguous, matches B): element pair g -> k = 16*half + 2g
#define KMAP(g, half) (((half) << 4) + ((g) << 1))

// generic __shared__ pointer -> 32-bit LDS byte address (flat LDS addr = low 32 bits)
__device__ __forceinline__ unsigned lds_off(const void* p) {
  return (unsigned)(uintptr_t)p;
}
// CDNA5 async copy: global -> LDS, 16 bytes per lane, tracked by ASYNCcnt
__device__ __forceinline__ void async_load_b128(unsigned lds, const void* gptr) {
  asm volatile("global_load_async_to_lds_b128 %0, %1, off"
               :: "v"(lds), "v"(gptr) : "memory");
}
__device__ __forceinline__ void wait_async0() {
#if __has_builtin(__builtin_amdgcn_s_wait_asynccnt)
  __builtin_amdgcn_s_wait_asynccnt(0);
#else
  asm volatile("s_wait_asynccnt 0x0" ::: "memory");
#endif
}

// ---------------------------------------------------------------- f32 -> bf16
__global__ void cvt_bf16(const float* __restrict__ src,
                         unsigned short* __restrict__ dst, size_t n4) {
  size_t i = blockIdx.x * (size_t)blockDim.x + threadIdx.x;
  size_t stride = (size_t)gridDim.x * blockDim.x;
  const float4* s4 = (const float4*)src;
  for (; i < n4; i += stride) {
    float4 f = s4[i];
    unsigned short r[4] = { f2bf(f.x), f2bf(f.y), f2bf(f.z), f2bf(f.w) };
    *(uint2*)(dst + 4 * i) = *(uint2*)r;
  }
}

// ------------------------------------------------- C = act(A[M,K] @ W[K,N] + b)
// ACT: 0 = none->bf16, 1 = elu+1->bf16, 2 = none->f32
// 128x128 block tile, 8 waves each 32x64; double-buffered LDS, async A copies.
template <int ACT>
__global__ __launch_bounds__(256) void gemm_bf16_nt(
    const unsigned short* __restrict__ A, const unsigned short* __restrict__ W,
    const float* __restrict__ bias, void* __restrict__ outp,
    int M, int N, int K) {
  __shared__ __attribute__((aligned(16))) unsigned short As[2][128 * 40]; // [m][k] stride 40 (16B rows, 20-bank stride)
  __shared__ __attribute__((aligned(16))) unsigned short Bs[2][128 * 34]; // [n][k] (W tile transposed)
  const int t = threadIdx.x;
  const int wave = t >> 5, lane = t & 31;
  const int half = lane >> 4, lr = lane & 15;
  const int m0 = blockIdx.y * 128, n0 = blockIdx.x * 128;
  const int wm = (wave & 3) * 32;           // wave's 32 rows
  const int wn = (wave >> 2) * 64;          // wave's 64 cols

  const int arow = t >> 1, aseg = (t & 1) << 4;   // A: 16 u16 per thread
  const int bk = t >> 3, bnseg = (t & 7) << 4;    // W: row bk, 16 cols

  const unsigned short* Ag = A + (size_t)(m0 + arow) * K + aseg;
  const unsigned short* Wg = W + (size_t)bk * N + n0 + bnseg;
  const unsigned aoff0 = lds_off(&As[0][arow * 40 + aseg]);
  const unsigned aoff1 = lds_off(&As[1][arow * 40 + aseg]);

  v8f acc[2][4] = {};

  // ---- prologue: chunk 0 into buffer 0
  async_load_b128(aoff0, Ag);
  async_load_b128(aoff0 + 16, Ag + 8);
  uint4 bx0 = ((const uint4*)Wg)[0];
  uint4 bx1 = ((const uint4*)Wg)[1];
  {
    unsigned short tmp[16];
    *(uint4*)&tmp[0] = bx0; *(uint4*)&tmp[8] = bx1;
#pragma unroll
    for (int i = 0; i < 16; ++i) Bs[0][(bnseg + i) * 34 + bk] = tmp[i];
  }
  wait_async0();
  __syncthreads();

  int p = 0;
  for (int k0 = 0; k0 < K; k0 += 32, p ^= 1) {
    const bool more = (k0 + 32 < K);
    if (more) {  // issue next-chunk prefetch into buffer p^1
      const unsigned short* Agn = Ag + (k0 + 32);
      const unsigned an = p ? aoff0 : aoff1;
      async_load_b128(an, Agn);
      async_load_b128(an + 16, Agn + 8);
      const unsigned short* Wgn = Wg + (size_t)(k0 + 32) * N;
      bx0 = ((const uint4*)Wgn)[0];
      bx1 = ((const uint4*)Wgn)[1];
    }
    // ---- compute on buffer p
    Frag16 a[2];
#pragma unroll
    for (int i = 0; i < 2; ++i) {
      const unsigned short* pa = &As[p][(wm + i * 16 + lr) * 40];
#pragma unroll
      for (int g = 0; g < 8; ++g) a[i].u[g] = *(const unsigned*)(pa + KMAP(g, half));
    }
#pragma unroll
    for (int j = 0; j < 4; ++j) {
      Frag16 bfr;
      const unsigned short* pb = &Bs[p][(wn + j * 16 + lr) * 34 + (half << 4)];
#pragma unroll
      for (int v = 0; v < 8; ++v) bfr.u[v] = *(const unsigned*)(pb + (v << 1));
#pragma unroll
      for (int i = 0; i < 2; ++i)
        acc[i][j] = __builtin_amdgcn_wmma_f32_16x16x32_bf16(
            false, a[i].v, false, bfr.v, (short)0, acc[i][j], false, false);
    }
    if (more) {  // publish prefetched B, wait for async A, flip
      unsigned short tmp[16];
      *(uint4*)&tmp[0] = bx0; *(uint4*)&tmp[8] = bx1;
#pragma unroll
      for (int i = 0; i < 16; ++i) Bs[p ^ 1][(bnseg + i) * 34 + bk] = tmp[i];
      wait_async0();
    }
    __syncthreads();
  }

  // ---- epilogue (branch-free, templated activation)
#pragma unroll
  for (int j = 0; j < 4; ++j) {
    const int col = n0 + wn + j * 16 + lr;
    const float bv = bias[col];
#pragma unroll
    for (int i = 0; i < 2; ++i) {
#pragma unroll
      for (int r = 0; r < 8; ++r) {
        const int row = m0 + wm + i * 16 + (half << 3) + r;
        float x = acc[i][j][r] + bv;
        if (ACT == 1) {
          const float xe = __expf(fminf(x, 0.0f));     // elu(x)+1, branchless
          x = (x > 0.0f) ? (x + 1.0f) : xe;
        }
        if (ACT == 2) ((float*)outp)[(size_t)row * N + col] = x;
        else          ((unsigned short*)outp)[(size_t)row * N + col] = f2bf(x);
      }
    }
  }
}

// ---- per (b,h,split): partial KV[e][d] = sum_s K[s,d]*V[s,e]; partial K_sum[d]
// K_sum comes free from an extra WMMA against an all-ones B fragment.
__global__ __launch_bounds__(256) void kv_reduce(
    const unsigned short* __restrict__ Kb, const unsigned short* __restrict__ Vb,
    float* __restrict__ pKV, float* __restrict__ pKs, int S, int H, int schunk) {
  __shared__ unsigned short Kt[64 * 34];    // [d][s]  (A = K^T)
  __shared__ unsigned short Vt[64 * 34];    // [e][s]  (B = V, column-major)
  const int bh = blockIdx.x;
  const int split = blockIdx.y;
  const int b = bh / H, h = bh % H;
  const int t = threadIdx.x;
  const int wave = t >> 5, lane = t & 31;
  const int half = lane >> 4, lr = lane & 15;
  const int wm = (wave & 3) * 16;           // d tile
  const int wn = (wave >> 2) * 32;          // two e tiles
  const size_t rowstride = (size_t)H * 64;
  const int sbeg = split * schunk;
  const int ls = t >> 3, dseg = (t & 7) << 3;
  const unsigned short* Kptr =
      Kb + (size_t)(b * S + sbeg + ls) * rowstride + h * 64 + dseg;
  const unsigned short* Vptr =
      Vb + (size_t)(b * S + sbeg + ls) * rowstride + h * 64 + dseg;

  v8f acc[2] = {};
  v8f accs = {};                            // K row sums
  Frag16 ones;
#pragma unroll
  for (int v = 0; v < 8; ++v) ones.u[v] = 0x3F803F80u;  // bf16 1.0 pair

  const int iters = schunk / 32;
  uint4 kx = *(const uint4*)Kptr;
  uint4 vx = *(const uint4*)Vptr;
  for (int it = 0; it < iters; ++it) {
    __syncthreads();
    {
      unsigned short tk[8], tv[8];
      *(uint4*)tk = kx; *(uint4*)tv = vx;
#pragma unroll
      for (int i = 0; i < 8; ++i) {
        Kt[(dseg + i) * 34 + ls] = tk[i];
        Vt[(dseg + i) * 34 + ls] = tv[i];
      }
    }
    if (it + 1 < iters) {  // register prefetch of next chunk (overlaps WMMA)
      kx = *(const uint4*)(Kptr + (size_t)(it + 1) * 32 * rowstride);
      vx = *(const uint4*)(Vptr + (size_t)(it + 1) * 32 * rowstride);
    }
    __syncthreads();
    Frag16 a;
    const unsigned short* pa = &Kt[(wm + lr) * 34];
#pragma unroll
    for (int g = 0; g < 8; ++g) a.u[g] = *(const unsigned*)(pa + KMAP(g, half));
#pragma unroll
    for (int j = 0; j < 2; ++j) {
      Frag16 bfr;
      const unsigned short* pb = &Vt[(wn + j * 16 + lr) * 34 + (half << 4)];
#pragma unroll
      for (int v = 0; v < 8; ++v) bfr.u[v] = *(const unsigned*)(pb + (v << 1));
      acc[j] = __builtin_amdgcn_wmma_f32_16x16x32_bf16(
          false, a.v, false, bfr.v, (short)0, acc[j], false, false);
    }
    accs = __builtin_amdgcn_wmma_f32_16x16x32_bf16(
        false, a.v, false, ones.v, (short)0, accs, false, false);
  }
  float* out = pKV + ((size_t)split * 64 + bh) * 4096;
#pragma unroll
  for (int j = 0; j < 2; ++j) {
    const int e = wn + j * 16 + lr;
#pragma unroll
    for (int r = 0; r < 8; ++r) {
      const int d = wm + (half << 3) + r;
      out[e * 64 + d] = acc[j][r];          // transposed f32 partial
    }
  }
  if (wave < 4 && lr == 0) {                // every column of accs holds the sum
#pragma unroll
    for (int r = 0; r < 8; ++r)
      pKs[((size_t)split * 64 + bh) * 64 + wm + (half << 3) + r] = accs[r];
  }
}

// ---- deterministic reduction of split partials -> bf16 KV^T and f32 K_sum
__global__ __launch_bounds__(256) void kv_finalize(
    const float* __restrict__ pKV, const float* __restrict__ pKs,
    unsigned short* __restrict__ KVt, float* __restrict__ Ksum, int nsplit) {
  const int bh = blockIdx.x;
  const int t = threadIdx.x;
  for (int idx = t; idx < 4096; idx += 256) {
    float s = 0.0f;
    for (int sp = 0; sp < nsplit; ++sp)
      s += pKV[((size_t)sp * 64 + bh) * 4096 + idx];
    KVt[(size_t)bh * 4096 + idx] = f2bf(s);
  }
  if (t < 64) {
    float s = 0.0f;
    for (int sp = 0; sp < nsplit; ++sp)
      s += pKs[((size_t)sp * 64 + bh) * 64 + t];
    Ksum[bh * 64 + t] = s;
  }
}

// ---------------- out[s,e] = (Q[s,:] @ KV[:,e]) / (Q[s,:].Ksum + 1e-6), per head
// norm computed by WMMA against a broadcast K_sum fragment: lands in the same
// row layout as the accumulators, so no LDS round-trip for the division.
__global__ __launch_bounds__(256) void apply_kv(
    const unsigned short* __restrict__ Qb, const unsigned short* __restrict__ KVt,
    const float* __restrict__ Ksum, unsigned short* __restrict__ Attn,
    int S, int H) {
  __shared__ unsigned short Qs[128 * 66];   // [s][d]
  __shared__ unsigned short Bs[64 * 66];    // [e][d]
  __shared__ float ksum_s[64];
  const int bh = blockIdx.x;
  const int s0 = blockIdx.y * 128;
  const int b = bh / H, h = bh % H;
  const int t = threadIdx.x;
  const int wave = t >> 5, lane = t & 31;
  const int half = lane >> 4, lr = lane & 15;
  const size_t rowstride = (size_t)H * 64;
  const unsigned short* Qbase = Qb + (size_t)(b * S + s0) * rowstride + (size_t)h * 64;

  {
    const int s = t >> 1, seg = (t & 1) << 5;
    const uint4* g = (const uint4*)(Qbase + (size_t)s * rowstride + seg);
    st8(&Qs[s * 66 + seg], g[0]);
    st8(&Qs[s * 66 + seg + 8], g[1]);
    st8(&Qs[s * 66 + seg + 16], g[2]);
    st8(&Qs[s * 66 + seg + 24], g[3]);
  }
  {
    const int e = t >> 2, dseg = (t & 3) << 4;
    const uint4* g = (const uint4*)(KVt + (size_t)bh * 4096 + e * 64 + dseg);
    st8(&Bs[e * 66 + dseg], g[0]);
    st8(&Bs[e * 66 + dseg + 8], g[1]);
  }
  if (t < 64) ksum_s[t] = Ksum[bh * 64 + t];
  __syncthreads();

  // broadcast K_sum into a B fragment (same value in every column)
  Frag16 bks[2];
#pragma unroll
  for (int c = 0; c < 2; ++c)
#pragma unroll
    for (int v = 0; v < 8; ++v) {
      const int k = (c << 5) + (half << 4) + (v << 1);
      const unsigned lo = f2bf(ksum_s[k]);
      const unsigned hi = f2bf(ksum_s[k + 1]);
      bks[c].u[v] = lo | (hi << 16);
    }

  v8f acc[4] = {};
  v8f nacc = {};
#pragma unroll
  for (int c = 0; c < 2; ++c) {
    Frag16 a;
    const unsigned short* pa = &Qs[(wave * 16 + lr) * 66 + (c << 5)];
#pragma unroll
    for (int g = 0; g < 8; ++g) a.u[g] = *(const unsigned*)(pa + KMAP(g, half));
#pragma unroll
    for (int j = 0; j < 4; ++j) {
      Frag16 bfr;
      const unsigned short* pb = &Bs[(j * 16 + lr) * 66 + (c << 5) + (half << 4)];
#pragma unroll
      for (int v = 0; v < 8; ++v) bfr.u[v] = *(const unsigned*)(pb + (v << 1));
      acc[j] = __builtin_amdgcn_wmma_f32_16x16x32_bf16(
          false, a.v, false, bfr.v, (short)0, acc[j], false, false);
    }
    nacc = __builtin_amdgcn_wmma_f32_16x16x32_bf16(
        false, a.v, false, bks[c].v, (short)0, nacc, false, false);
  }
#pragma unroll
  for (int r = 0; r < 8; ++r) {
    const int sl = wave * 16 + (half << 3) + r;
    const float inv = 1.0f / (nacc[r] + 1e-6f);
    unsigned short* orow = Attn + (size_t)(b * S + s0 + sl) * rowstride + h * 64;
#pragma unroll
    for (int j = 0; j < 4; ++j) orow[j * 16 + lr] = f2bf(acc[j][r] * inv);
  }
}

extern "C" void kernel_launch(void* const* d_in, const int* in_sizes, int n_in,
                              void* d_out, int out_size, void* d_ws, size_t ws_size,
                              hipStream_t stream) {
  (void)in_sizes; (void)n_in; (void)out_size; (void)ws_size;
  const int B = 4, S = 4096, D = 1024, H = 16, NSPLIT = 8;
  const float* query     = (const float*)d_in[0];
  const float* key_value = (const float*)d_in[1];
  const float* Wq = (const float*)d_in[2];
  const float* bq = (const float*)d_in[3];
  const float* Wk = (const float*)d_in[4];
  const float* bk = (const float*)d_in[5];
  const float* Wv = (const float*)d_in[6];
  const float* bv = (const float*)d_in[7];
  const float* Wo = (const float*)d_in[8];
  const float* bo = (const float*)d_in[9];

  const size_t SQD = (size_t)B * S * D;   // 16,777,216
  const size_t WSZ = (size_t)D * D;
  const size_t BH  = (size_t)B * H;       // 64
  char* p = (char*)d_ws;
  unsigned short* qbf  = (unsigned short*)p; p += SQD * 2;
  unsigned short* kvbf = (unsigned short*)p; p += SQD * 2;
  unsigned short* wqb  = (unsigned short*)p; p += WSZ * 2;
  unsigned short* wkb  = (unsigned short*)p; p += WSZ * 2;
  unsigned short* wvb  = (unsigned short*)p; p += WSZ * 2;
  unsigned short* wob  = (unsigned short*)p; p += WSZ * 2;
  unsigned short* Qbf  = (unsigned short*)p; p += SQD * 2;
  unsigned short* Kbf  = (unsigned short*)p; p += SQD * 2;
  unsigned short* Vbf  = (unsigned short*)p; p += SQD * 2;
  unsigned short* KVt  = (unsigned short*)p; p += BH * 64 * 64 * 2;
  float* Ksum          = (float*)p;          p += BH * 64 * 4;
  float* pKV           = (float*)p;          p += (size_t)NSPLIT * BH * 64 * 64 * 4;
  float* pKs           = (float*)p;          p += (size_t)NSPLIT * BH * 64 * 4;
  unsigned short* attn = qbf;  // qbf is dead after the Q projection GEMM

  auto cvt = [&](const float* s, unsigned short* d, size_t n) {
    size_t n4 = n / 4;
    int grid = (int)((n4 + 255) / 256);
    if (grid > 8192) grid = 8192;
    cvt_bf16<<<grid, 256, 0, stream>>>(s, d, n4);
  };
  cvt(query, qbf, SQD);
  cvt(key_value, kvbf, SQD);
  cvt(Wq, wqb, WSZ);
  cvt(Wk, wkb, WSZ);
  cvt(Wv, wvb, WSZ);
  cvt(Wo, wob, WSZ);

  dim3 gg(D / 128, (B * S) / 128);  // (8, 128)
  gemm_bf16_nt<1><<<gg, 256, 0, stream>>>(qbf, wqb, bq, Qbf, B * S, D, D);
  gemm_bf16_nt<1><<<gg, 256, 0, stream>>>(kvbf, wkb, bk, Kbf, B * S, D, D);
  gemm_bf16_nt<0><<<gg, 256, 0, stream>>>(kvbf, wvb, bv, Vbf, B * S, D, D);

  kv_reduce<<<dim3((int)BH, NSPLIT), 256, 0, stream>>>(Kbf, Vbf, pKV, pKs, S, H, S / NSPLIT);
  kv_finalize<<<(int)BH, 256, 0, stream>>>(pKV, pKs, KVt, Ksum, NSPLIT);
  apply_kv<<<dim3((int)BH, S / 128), 256, 0, stream>>>(Qbf, KVt, Ksum, attn, S, H);

  gemm_bf16_nt<2><<<gg, 256, 0, stream>>>(attn, wob, bo, d_out, B * S, D, D);
}
